// SBCoreMiniLM_28587302323148
// MI455X (gfx1250) — compile-verified, matrix operand order checked by reading, not run on hardware
//
#include <hip/hip_runtime.h>
#include <hip/hip_bf16.h>

typedef __attribute__((ext_vector_type(16))) _Float16 v16h;
typedef __attribute__((ext_vector_type(8)))  _Float16 v8h;
typedef __attribute__((ext_vector_type(8)))  float    v8f;

#define BB   32768
#define DD   128
#define NN   8
#define ROWS 16
#define TEMP_INV 4.0f   // 1/0.25

__device__ __forceinline__ float wsum(float v) {
#pragma unroll
  for (int o = 16; o > 0; o >>= 1) v += __shfl_xor(v, o, 32);
  return v;
}
__device__ __forceinline__ float sigmoidf_(float x) { return 1.0f / (1.0f + expf(-x)); }
__device__ __forceinline__ float clamp01(float x) { return fminf(fmaxf(x, 0.0f), 1.0f); }

// A fragment: 16x32 f16, lane l holds row m=l&15; k-halves selected by h=l>>4.
// Elements a[0..7]  = k in [32c+8h, 32c+8h+8)
//          a[8..15] = k in [32c+16+8h, ...)
__device__ __forceinline__ v16h load_a(const float* __restrict__ sig,
                                       const float* __restrict__ oth,
                                       int Rbase, int lane, int c) {
  const int m = lane & 15, h = lane >> 4;
  const float* src = (c < 4) ? sig : oth;
  const int col = 32 * (c & 3);
  const float* p = src + (size_t)(Rbase + m) * DD + col + 8 * h;
  const float4 f0 = *(const float4*)(p);
  const float4 f1 = *(const float4*)(p + 4);
  const float4 f2 = *(const float4*)(p + 16);
  const float4 f3 = *(const float4*)(p + 20);
  v16h a;
  a[0]=(_Float16)f0.x;  a[1]=(_Float16)f0.y;  a[2]=(_Float16)f0.z;  a[3]=(_Float16)f0.w;
  a[4]=(_Float16)f1.x;  a[5]=(_Float16)f1.y;  a[6]=(_Float16)f1.z;  a[7]=(_Float16)f1.w;
  a[8]=(_Float16)f2.x;  a[9]=(_Float16)f2.y;  a[10]=(_Float16)f2.z; a[11]=(_Float16)f2.w;
  a[12]=(_Float16)f3.x; a[13]=(_Float16)f3.y; a[14]=(_Float16)f3.z; a[15]=(_Float16)f3.w;
  return a;
}

// B fragment: 32x16 f16 from LDS chunk stored as [n][k] (k contiguous, 32 per n).
__device__ __forceinline__ v16h load_b(const _Float16* __restrict__ lwm, int lane, int t) {
  const int nl = lane & 15, h = lane >> 4;
  const _Float16* p = lwm + (size_t)(16 * t + nl) * 32 + 8 * h;
  const v8h b0 = *(const v8h*)p;
  const v8h b1 = *(const v8h*)(p + 16);
  return __builtin_shufflevector(b0, b1, 0,1,2,3,4,5,6,7,8,9,10,11,12,13,14,15);
}

__global__ __launch_bounds__(128) void memcell_kernel(
    const float* __restrict__ signal, const float* __restrict__ previous,
    const float* __restrict__ hidden, const float* __restrict__ ekeys,
    const float* __restrict__ evals, const float* __restrict__ estr,
    const float* __restrict__ eage,
    const float* __restrict__ w_rq,  const float* __restrict__ w_rmix,
    const float* __restrict__ b_rmix, const float* __restrict__ w_key,
    const float* __restrict__ b_key, const float* __restrict__ w_val,
    const float* __restrict__ b_val, const float* __restrict__ w_wr,
    const float* __restrict__ b_wr,  const float* __restrict__ w_pe,
    const float* __restrict__ b_pe,  const float* __restrict__ w_mg,
    const float* __restrict__ b_mg,
    float* __restrict__ out_mr, float* __restrict__ out_k,
    float* __restrict__ out_v,  float* __restrict__ out_s,
    float* __restrict__ out_a) {
  __shared__ alignas(16) _Float16 lw[3][DD][32];   // transposed f16 weight K-chunk: 24 KB
  __shared__ float qs[ROWS][DD];                   // raw query rows
  __shared__ float cks[ROWS][DD];                  // tanh candidate key
  __shared__ float cvs[ROWS][DD];                  // tanh candidate value
  __shared__ float gl[ROWS][4];                    // gate logits (no bias)

  const int tid  = threadIdx.x;
  const int lane = tid & 31;
  const int wave = tid >> 5;
  const int Rbase = blockIdx.x * ROWS;

  v8f acc[8];
#pragma unroll
  for (int t = 0; t < 8; ++t)
#pragma unroll
    for (int v = 0; v < 8; ++v) acc[t][v] = 0.0f;

  // ---- GEMM phase: 8 K-chunks of 32 ----
  for (int c = 0; c < 8; ++c) {
    __syncthreads();
    // cooperative stage: W[32c+k][n] (f32, coalesced) -> lw[mat][n][k] (f16)
    for (int i = tid; i < 3 * 32 * DD; i += 128) {
      const int mat = i >> 12;
      const int rem = i & 4095;
      const int k = rem >> 7;
      const int n = rem & 127;
      const float* wp = (mat == 0) ? w_rq : (mat == 1) ? w_key : w_val;
      lw[mat][n][k] = (_Float16)wp[(size_t)(32 * c + k) * DD + n];
    }
    __syncthreads();

    if (wave < 3) {
      const float* oth = (wave == 0) ? previous : hidden;
      const v16h a = load_a(signal, oth, Rbase, lane, c);
#pragma unroll
      for (int t = 0; t < 8; ++t) {
        const v16h b = load_b(&lw[wave][0][0], lane, t);
        acc[t] = __builtin_amdgcn_wmma_f32_16x16x32_f16(
            false, a, false, b, (short)0, acc[t], false, false);
      }
    } else if (c == 0) {
      // wave 3: gate logits, 2 (row,gate) pairs per lane, 256-length dots
#pragma unroll
      for (int pp = 0; pp < 2; ++pp) {
        const int p = lane + 32 * pp;
        const int row = p >> 2, g = p & 3;
        const int R = Rbase + row;
        const float* wg = (g == 0) ? w_rmix : (g == 1) ? w_wr : (g == 2) ? w_pe : w_mg;
        const float* x2 = (g == 0) ? previous : hidden;
        float s = 0.0f;
        for (int k = 0; k < DD; ++k) s += signal[(size_t)R * DD + k] * wg[k];
        for (int k = 0; k < DD; ++k) s += x2[(size_t)R * DD + k] * wg[DD + k];
        gl[row][g] = s;
      }
    }
  }

  // ---- write GEMM results to LDS (D layout: lane n=l&15; VGPR v -> m = v + 8*(l>>4)) ----
  if (wave < 3) {
    const int nl = lane & 15, mh = (lane >> 4) * 8;
#pragma unroll
    for (int t = 0; t < 8; ++t) {
      const int n = 16 * t + nl;
      const float bias = (wave == 1) ? b_key[n] : (wave == 2) ? b_val[n] : 0.0f;
#pragma unroll
      for (int v = 0; v < 8; ++v) {
        const float x = acc[t][v] + bias;
        if (wave == 0)      qs[mh + v][n]  = x;
        else if (wave == 1) cks[mh + v][n] = tanhf(x);
        else                cvs[mh + v][n] = tanhf(x);
      }
    }
  }
  __syncthreads();

  // ---- epilogue: each wave handles 4 rows; 32 lanes x 4 columns each ----
  for (int rr = 0; rr < 4; ++rr) {
    const int r = wave * 4 + rr;
    const int R = Rbase + r;

    float q4[4], ck4[4], cv4[4];
    float qss = 0.0f, css = 0.0f;
#pragma unroll
    for (int j = 0; j < 4; ++j) {
      const int d = lane + 32 * j;
      q4[j] = qs[r][d]; ck4[j] = cks[r][d]; cv4[j] = cvs[r][d];
      qss += q4[j] * q4[j]; css += ck4[j] * ck4[j];
    }
    qss = wsum(qss); css = wsum(css);
    const float qinv = 1.0f / fmaxf(sqrtf(qss), 1e-6f);
    const float cinv = 1.0f / fmaxf(sqrtf(css), 1e-6f);

    float kreg[NN][4], vreg[NN][4], simc[NN], simk[NN];
#pragma unroll
    for (int n = 0; n < NN; ++n) {
      const size_t base = ((size_t)R * NN + n) * DD;
      float dq = 0.0f, dc = 0.0f, kk = 0.0f;
#pragma unroll
      for (int j = 0; j < 4; ++j) {
        const float kv = ekeys[base + lane + 32 * j];
        const float vv = evals[base + lane + 32 * j];
        kreg[n][j] = kv; vreg[n][j] = vv;
        dq += q4[j] * kv; dc += ck4[j] * kv; kk += kv * kv;
      }
      dq = wsum(dq); dc = wsum(dc); kk = wsum(kk);
      const float kinv = 1.0f / fmaxf(sqrtf(kk), 1e-6f);
      simc[n] = dq * qinv * kinv;
      simk[n] = dc * cinv * kinv;
    }

    // content softmax
    float mx = simc[0];
#pragma unroll
    for (int n = 1; n < NN; ++n) mx = fmaxf(mx, simc[n]);
    float wc[NN], se = 0.0f;
#pragma unroll
    for (int n = 0; n < NN; ++n) { wc[n] = expf((simc[n] - mx) * TEMP_INV); se += wc[n]; }
    const float sei = 1.0f / se;
#pragma unroll
    for (int n = 0; n < NN; ++n) wc[n] *= sei;

    // persistent softmax
    float str_[NN], age_[NN], ps[NN];
#pragma unroll
    for (int n = 0; n < NN; ++n) {
      str_[n] = estr[(size_t)R * NN + n];
      age_[n] = eage[(size_t)R * NN + n];
      ps[n] = 2.4f * str_[n] + 0.6f * (1.0f - age_[n]);
    }
    float pmx = ps[0];
#pragma unroll
    for (int n = 1; n < NN; ++n) pmx = fmaxf(pmx, ps[n]);
    float wp[NN], pse = 0.0f;
#pragma unroll
    for (int n = 0; n < NN; ++n) { wp[n] = expf((ps[n] - pmx) * TEMP_INV); pse += wp[n]; }
    const float psei = 1.0f / pse;
#pragma unroll
    for (int n = 0; n < NN; ++n) wp[n] *= psei;

    // reads
    float cr[4] = {0, 0, 0, 0}, pr[4] = {0, 0, 0, 0};
#pragma unroll
    for (int n = 0; n < NN; ++n)
#pragma unroll
      for (int j = 0; j < 4; ++j) {
        cr[j] += wc[n] * vreg[n][j];
        pr[j] += wp[n] * vreg[n][j];
      }

    const float rmix = sigmoidf_(gl[r][0] + b_rmix[0]);
    const float wstr = sigmoidf_(gl[r][1] + b_wr[0]);
    const float pers = sigmoidf_(gl[r][2] + b_pe[0]);
#pragma unroll
    for (int j = 0; j < 4; ++j)
      out_mr[(size_t)R * DD + lane + 32 * j] = rmix * cr[j] + (1.0f - rmix) * pr[j];

    // top-3 of simk (stable: earlier index wins ties)
    float s0 = -1e30f, s1 = -1e30f, s2 = -1e30f;
    int i0 = 0, i1 = 0, i2 = 0;
#pragma unroll
    for (int n = 0; n < NN; ++n) {
      const float v = simk[n];
      if (v > s0)      { s2 = s1; i2 = i1; s1 = s0; i1 = i0; s0 = v; i0 = n; }
      else if (v > s1) { s2 = s1; i2 = i1; s1 = v; i1 = n; }
      else if (v > s2) { s2 = v; i2 = n; }
    }

    const float mpref = sigmoidf_(gl[r][3] + b_mg[0] + 2.6f * s0);
    const bool full  = (s0 > 0.78f) && (mpref >= 0.55f);
    const bool multi = full && (s1 > 0.68f);
    const bool part  = (!multi) && (s0 > 0.64f) && (s1 > 0.52f);

    // replace argmax
    int ridx = 0; float rbest = -1e30f;
#pragma unroll
    for (int n = 0; n < NN; ++n) {
      const float rs = 1.2f * age_[n] + (1.0f - str_[n]) + 0.5f * (1.0f - simk[n]);
      if (rs > rbest) { rbest = rs; ridx = n; }
    }

    float tw[NN];
#pragma unroll
    for (int n = 0; n < NN; ++n) tw[n] = 0.0f;
    tw[ridx] = 1.0f;
    if (full) {
#pragma unroll
      for (int n = 0; n < NN; ++n) tw[n] = 0.0f;
      tw[i0] = 1.0f;
    }
    if (part) {
#pragma unroll
      for (int n = 0; n < NN; ++n) tw[n] = 0.0f;
      const float e1 = expf((s1 - s0) * TEMP_INV);
      const float d = 1.0f / (1.0f + e1);
      tw[i0] = d; tw[i1] = e1 * d;
    }
    if (multi) {
#pragma unroll
      for (int n = 0; n < NN; ++n) tw[n] = 0.0f;
      const float e1 = expf((s1 - s0) * TEMP_INV);
      const float e2 = expf((s2 - s0) * TEMP_INV);
      const float d = 1.0f / (1.0f + e1 + e2);
      tw[i0] = d; tw[i1] = e1 * d; tw[i2] = e2 * d;
    }

    const float nov = clamp01(1.0f - s0);
    const float scale = multi ? (0.16f + 0.52f * wstr)
                      : part  ? (0.18f + 0.62f * wstr)
                              : (0.20f + 0.80f * wstr);
    const bool  ml   = full || part || multi;
    const float kmix = ml ? (0.28f + 0.24f * pers) : (0.78f + 0.16f * pers);
    const float vmix = ml ? (0.42f + 0.28f * pers) : (0.82f + 0.12f * pers);
    const float bco  = 0.45f + 0.35f * pers + 0.45f * nov + 0.25f * wstr;
    const float ons  = scale * (0.55f + 0.45f * nov);

#pragma unroll
    for (int n = 0; n < NN; ++n) {
      const float ow = tw[n] * ons;
      const size_t base = ((size_t)R * NN + n) * DD;
      const float ka = ow * kmix, va = ow * vmix;
#pragma unroll
      for (int j = 0; j < 4; ++j) {
        const size_t idx = base + lane + 32 * j;
        out_k[idx] = kreg[n][j] + ka * (ck4[j] - kreg[n][j]);
        out_v[idx] = vreg[n][j] + va * (cv4[j] - vreg[n][j]);
      }
      if (lane == n) {
        out_s[(size_t)R * NN + n] = clamp01(str_[n] * 0.99f + ow * bco);
        out_a[(size_t)R * NN + n] = clamp01((age_[n] + 0.02f) * (1.0f - ow));
      }
    }
  }
}

extern "C" void kernel_launch(void* const* d_in, const int* in_sizes, int n_in,
                              void* d_out, int out_size, void* d_ws, size_t ws_size,
                              hipStream_t stream) {
  (void)in_sizes; (void)n_in; (void)out_size; (void)d_ws; (void)ws_size;
  const float* signal   = (const float*)d_in[0];
  const float* previous = (const float*)d_in[1];
  const float* hidden   = (const float*)d_in[2];
  const float* ekeys    = (const float*)d_in[3];
  const float* evals    = (const float*)d_in[4];
  const float* estr     = (const float*)d_in[5];
  const float* eage     = (const float*)d_in[6];
  const float* w_rq     = (const float*)d_in[7];
  const float* w_rmix   = (const float*)d_in[8];
  const float* b_rmix   = (const float*)d_in[9];
  const float* w_key    = (const float*)d_in[10];
  const float* b_key    = (const float*)d_in[11];
  const float* w_val    = (const float*)d_in[12];
  const float* b_val    = (const float*)d_in[13];
  const float* w_wr     = (const float*)d_in[14];
  const float* b_wr     = (const float*)d_in[15];
  const float* w_pe     = (const float*)d_in[16];
  const float* b_pe     = (const float*)d_in[17];
  const float* w_mg     = (const float*)d_in[18];
  const float* b_mg     = (const float*)d_in[19];

  float* out_mr = (float*)d_out;
  float* out_k  = out_mr + (size_t)BB * DD;
  float* out_v  = out_k + (size_t)BB * NN * DD;
  float* out_s  = out_v + (size_t)BB * NN * DD;
  float* out_a  = out_s + (size_t)BB * NN;

  hipLaunchKernelGGL(memcell_kernel, dim3(BB / ROWS), dim3(128), 0, stream,
                     signal, previous, hidden, ekeys, evals, estr, eage,
                     w_rq, w_rmix, b_rmix, w_key, b_key, w_val, b_val,
                     w_wr, b_wr, w_pe, b_pe, w_mg, b_mg,
                     out_mr, out_k, out_v, out_s, out_a);
}
